// RFSQDraftModelWithProjection_24979529794189
// MI455X (gfx1250) — compile-verified, matrix operand order checked by reading, not run on hardware
//
#include <hip/hip_runtime.h>
#include <hip/hip_bf16.h>
#include <math.h>

// ---------------------------------------------------------------------------
// CDNA5 (gfx1250) bf16 WMMA GEMM pipeline for the seq-len-1 decoder + heads.
// f32 in / f32 out; operands converted to bf16 while staging into LDS,
// accumulation in f32 via v_wmma_f32_16x16x32_bf16. Double-buffered LDS,
// 128x128 block tile, 32x64 wave tile (2x4 WMMA accumulators).
// ---------------------------------------------------------------------------

typedef __bf16 bf16_t;
typedef __attribute__((ext_vector_type(16))) __bf16 v16bf;
typedef __attribute__((ext_vector_type(8)))  __bf16 v8bf;
typedef __attribute__((ext_vector_type(8)))  float  v8f;

#define BM 128
#define BN 128
#define BK 32
#define LDK 40   // padded LDS row pitch (elements); 80B keeps 16B alignment

__device__ __forceinline__ v8f wmma_bf16f32(v16bf a, v16bf b, v8f c) {
  // 8 args: (neg_a, A, neg_b, B, c_mod, C, reuse_a, reuse_b)
  return __builtin_amdgcn_wmma_f32_16x16x32_bf16(false, a, false, b, (short)0, c,
                                                 false, false);
}

// Per-lane fragment: 8 contiguous bf16 at kbase and 8 at kbase+16
// (16-bit A-matrix 16x32 VGPR layout; B tile is stored N-major in LDS so its
// fragment addressing is identical with row := column).
__device__ __forceinline__ v16bf frag_from_lds(const bf16_t* rowp, int kbase) {
  v8bf lo = *(const v8bf*)(rowp + kbase);
  v8bf hi = *(const v8bf*)(rowp + kbase + 16);
  return __builtin_shufflevector(lo, hi, 0, 1, 2, 3, 4, 5, 6, 7,
                                         8, 9, 10, 11, 12, 13, 14, 15);
}

__device__ __forceinline__ float gelu_exact(float x) {
  return 0.5f * x * (1.0f + erff(x * 0.70710678118654752440f));
}

__device__ __forceinline__ void store_tile(v8f a, float* __restrict__ C,
                                           const float* __restrict__ bias,
                                           const float* __restrict__ bias2,
                                           const float* __restrict__ res,
                                           int ldc, int ldr, int rowb, int col,
                                           int doGelu) {
  float bv = 0.0f;
  if (bias)  bv += bias[col];
  if (bias2) bv += bias2[col];
#pragma unroll
  for (int r = 0; r < 8; ++r) {
    float v = a[r] + bv;
    if (res) v += res[(size_t)(rowb + r) * ldr + col];
    if (doGelu) v = gelu_exact(v);
    C[(size_t)(rowb + r) * ldc + col] = v;
  }
}

__global__ __launch_bounds__(256)
void wmma_gemm_bf16(const float* __restrict__ A, const float* __restrict__ B,
                    float* __restrict__ C,
                    const float* __restrict__ bias,
                    const float* __restrict__ bias2,
                    const float* __restrict__ res,
                    int M, int N, int K, int lda, int ldb, int ldc, int ldr,
                    long long sA, long long sB, long long sC,
                    long long sBias, long long sRes, int doGelu) {
  __shared__ __align__(16) bf16_t lA[2][BM * LDK];  // 2 x 10 KB
  __shared__ __align__(16) bf16_t lB[2][BN * LDK];  // 2 x 10 KB

  const int tid  = threadIdx.x;
  const int lane = tid & 31;
  const int wave = tid >> 5;
  const int bz   = blockIdx.z;
  const int m0   = blockIdx.y * BM;
  const int n0   = blockIdx.x * BN;

  A += (long long)bz * sA;
  B += (long long)bz * sB;
  C += (long long)bz * sC;
  if (bias) bias += (long long)bz * sBias;
  if (res)  res  += (long long)bz * sRes;

  const int wm = (wave & 3) * 32;   // 4 waves along M -> 128 rows
  const int wn = (wave >> 2) * 64;  // 2 waves along N -> 128 cols
  const int fr = lane & 15;         // fragment row (A) / col (B)
  const int kb = (lane >> 4) * 8;   // K half select per lane group

  // staging thread mapping (computed once)
  const int a_r0 = (tid * 8) >> 5;          // A rows for it=0 / it=1
  const int a_c0 = (tid * 8) & 31;
  const int a_r1 = (2048 + tid * 8) >> 5;
  const int a_c1 = (2048 + tid * 8) & 31;
  const int b_nn = tid & 127;               // B column (contiguous across lanes)
  const int b_k0 = (tid >> 7) * 8;          // K start for it=0 (it=1 adds 16)

  float4 ra[4];     // staged A regs (2 iters x 2 float4)
  float  rb[16];    // staged B regs (2 iters x 8 strided dwords)

  auto stage_load = [&](int k0) {
    const float4* pa0 = (const float4*)(A + (size_t)(m0 + a_r0) * lda + k0 + a_c0);
    ra[0] = pa0[0]; ra[1] = pa0[1];
    const float4* pa1 = (const float4*)(A + (size_t)(m0 + a_r1) * lda + k0 + a_c1);
    ra[2] = pa1[0]; ra[3] = pa1[1];
#pragma unroll
    for (int it = 0; it < 2; ++it) {
      const float* pb = B + (size_t)(k0 + b_k0 + it * 16) * ldb + n0 + b_nn;
#pragma unroll
      for (int j = 0; j < 8; ++j) rb[it * 8 + j] = pb[(size_t)j * ldb];
    }
  };
  auto stage_store = [&](bf16_t* dA, bf16_t* dB) {
    v8bf p0 = { (bf16_t)ra[0].x, (bf16_t)ra[0].y, (bf16_t)ra[0].z, (bf16_t)ra[0].w,
                (bf16_t)ra[1].x, (bf16_t)ra[1].y, (bf16_t)ra[1].z, (bf16_t)ra[1].w };
    *(v8bf*)(&dA[a_r0 * LDK + a_c0]) = p0;
    v8bf p1 = { (bf16_t)ra[2].x, (bf16_t)ra[2].y, (bf16_t)ra[2].z, (bf16_t)ra[2].w,
                (bf16_t)ra[3].x, (bf16_t)ra[3].y, (bf16_t)ra[3].z, (bf16_t)ra[3].w };
    *(v8bf*)(&dA[a_r1 * LDK + a_c1]) = p1;
#pragma unroll
    for (int it = 0; it < 2; ++it) {
      v8bf q = { (bf16_t)rb[it * 8 + 0], (bf16_t)rb[it * 8 + 1],
                 (bf16_t)rb[it * 8 + 2], (bf16_t)rb[it * 8 + 3],
                 (bf16_t)rb[it * 8 + 4], (bf16_t)rb[it * 8 + 5],
                 (bf16_t)rb[it * 8 + 6], (bf16_t)rb[it * 8 + 7] };
      *(v8bf*)(&dB[b_nn * LDK + b_k0 + it * 16]) = q;
    }
  };

  v8f acc[2][4] = {};

  // prologue: stage first K tile into buffer 0
  stage_load(0);
  stage_store(lA[0], lB[0]);
  int p = 0;

  for (int k0 = 0; k0 < K; k0 += BK) {
    __syncthreads();                    // buffer p ready for all waves
    const bool more = (k0 + BK) < K;
    if (more) stage_load(k0 + BK);      // global loads in flight over the WMMAs

    const bf16_t* cA = lA[p];
    const bf16_t* cB = lB[p];
    v16bf a0 = frag_from_lds(cA + (wm + fr) * LDK, kb);
    v16bf a1 = frag_from_lds(cA + (wm + 16 + fr) * LDK, kb);
#pragma unroll
    for (int j = 0; j < 4; ++j) {
      v16bf bf = frag_from_lds(cB + (wn + 16 * j + fr) * LDK, kb);
      acc[0][j] = wmma_bf16f32(a0, bf, acc[0][j]);
      acc[1][j] = wmma_bf16f32(a1, bf, acc[1][j]);
    }

    if (more) stage_store(lA[p ^ 1], lB[p ^ 1]);  // cvt + ds_store other buffer
    p ^= 1;
  }

  // ---- epilogue: bias (+bias2) + residual + optional GELU, f32 store -----
  const int rsel = 8 * (lane >> 4);
  const int csel = lane & 15;
#pragma unroll
  for (int i = 0; i < 2; ++i)
#pragma unroll
    for (int j = 0; j < 4; ++j)
      store_tile(acc[i][j], C, bias, bias2, res, ldc, ldr,
                 m0 + wm + 16 * i + rsel, n0 + wn + 16 * j + csel, doGelu);
}

// ---------------------------------------------------------------------------
// Row LayerNorm: one 256-thread block per row. rowsPerBatch selects the
// gamma/beta slice for the per-head LN (gamma stride == n).
// ---------------------------------------------------------------------------
__global__ __launch_bounds__(256)
void layernorm_kernel(const float* __restrict__ x, float* __restrict__ y,
                      const float* __restrict__ g, const float* __restrict__ b,
                      int n, int rowsPerBatch) {
  __shared__ float s1[256];
  __shared__ float s2[256];
  const int row = blockIdx.x;
  const int t   = threadIdx.x;
  const float* xr = x + (size_t)row * n;
  float a = 0.f, q = 0.f;
  for (int i = t; i < n; i += 256) {
    float v = xr[i];
    a += v;
    q += v * v;
  }
  s1[t] = a; s2[t] = q;
  __syncthreads();
  for (int s = 128; s > 0; s >>= 1) {
    if (t < s) { s1[t] += s1[t + s]; s2[t] += s2[t + s]; }
    __syncthreads();
  }
  const float mean = s1[0] / (float)n;
  const float var  = s2[0] / (float)n - mean * mean;
  const float rstd = rsqrtf(var + 1e-5f);
  const int batch = row / rowsPerBatch;
  const float* gb = g + (size_t)batch * n;
  const float* bb = b + (size_t)batch * n;
  float* yr = y + (size_t)row * n;
  for (int i = t; i < n; i += 256)
    yr[i] = (xr[i] - mean) * rstd * gb[i] + bb[i];
}

// ---------------------------------------------------------------------------
extern "C" void kernel_launch(void* const* d_in, const int* in_sizes, int n_in,
                              void* d_out, int out_size, void* d_ws, size_t ws_size,
                              hipStream_t stream) {
  (void)in_sizes; (void)n_in; (void)out_size; (void)ws_size;

  const float* x       = (const float*)d_in[0];
  const float* w_in    = (const float*)d_in[1];
  const float* b_in    = (const float*)d_in[2];
  const float* pos     = (const float*)d_in[3];
  const float* ln1_g   = (const float*)d_in[4];
  const float* ln1_b   = (const float*)d_in[5];
  const float* wv_sa   = (const float*)d_in[6];
  const float* bv_sa   = (const float*)d_in[7];
  const float* wo_sa   = (const float*)d_in[8];
  const float* bo_sa   = (const float*)d_in[9];
  // d_in[10], d_in[11]: ln2 — mathematically unused (softmax over 1 element)
  const float* wv_ca   = (const float*)d_in[12];
  const float* bv_ca   = (const float*)d_in[13];
  const float* wo_ca   = (const float*)d_in[14];
  const float* bo_ca   = (const float*)d_in[15];
  const float* ln3_g   = (const float*)d_in[16];
  const float* ln3_b   = (const float*)d_in[17];
  const float* w_ff1   = (const float*)d_in[18];
  const float* b_ff1   = (const float*)d_in[19];
  const float* w_ff2   = (const float*)d_in[20];
  const float* b_ff2   = (const float*)d_in[21];
  const float* lnout_g = (const float*)d_in[22];
  const float* lnout_b = (const float*)d_in[23];
  const float* wh1     = (const float*)d_in[24];
  const float* bh1     = (const float*)d_in[25];
  const float* lnh_g   = (const float*)d_in[26];
  const float* lnh_b   = (const float*)d_in[27];
  const float* wh2     = (const float*)d_in[28];
  const float* bh2     = (const float*)d_in[29];
  float* out = (float*)d_out;

  constexpr int Bn = 8192, DIN = 4096, D = 512, F = 2048, NH = 3, HD = 256, OUTD = 896;

  // bump allocator over workspace
  char* wsp = (char*)d_ws;
  auto alloc = [&](size_t bytes) -> float* {
    char* p = wsp;
    wsp += (bytes + 255) & ~(size_t)255;
    return (float*)p;
  };
  float* buf_mem = alloc((size_t)Bn * D * 4);   // h / mem, later h4
  float* buf_t   = alloc((size_t)Bn * D * 4);   // LN outputs (t1/t3/ho)
  float* buf_u   = alloc((size_t)Bn * D * 4);   // V projections
  float* buf_h2  = alloc((size_t)Bn * D * 4);
  float* buf_h3  = alloc((size_t)Bn * D * 4);
  float* buf_f   = alloc((size_t)Bn * F * 4);   // FF intermediate; reused for heads
  float* buf_z   = buf_f;                       // [NH][Bn][HD] (buf_f is free by then)
  float* buf_zn  = buf_f + (size_t)NH * Bn * HD;

  auto gemm = [&](const float* A, const float* B, float* C, const float* bias,
                  const float* bias2, const float* res, int M, int N, int K,
                  int lda, int ldb, int ldc, int ldr, int batch,
                  long long sA, long long sB, long long sC, long long sBias,
                  long long sRes, int gelu) {
    dim3 grid(N / BN, M / BM, batch);
    wmma_gemm_bf16<<<grid, 256, 0, stream>>>(A, B, C, bias, bias2, res, M, N, K,
                                             lda, ldb, ldc, ldr, sA, sB, sC,
                                             sBias, sRes, gelu);
  };
  auto ln = [&](const float* xp, float* yp, const float* g, const float* b,
                int n, int rows, int rowsPerBatch) {
    layernorm_kernel<<<dim3(rows), 256, 0, stream>>>(xp, yp, g, b, n, rowsPerBatch);
  };

  // 1) h = x @ w_in + b_in + pos          [Bn, D]
  gemm(x, w_in, buf_mem, b_in, pos, nullptr, Bn, D, DIN, DIN, D, D, 0, 1, 0, 0, 0, 0, 0, 0);
  // 2) t1 = LN1(h)
  ln(buf_mem, buf_t, ln1_g, ln1_b, D, Bn, Bn);
  // 3) u = t1 @ wv_sa + bv_sa
  gemm(buf_t, wv_sa, buf_u, bv_sa, nullptr, nullptr, Bn, D, D, D, D, D, 0, 1, 0, 0, 0, 0, 0, 0);
  // 4) h2 = h + u @ wo_sa + bo_sa
  gemm(buf_u, wo_sa, buf_h2, bo_sa, nullptr, buf_mem, Bn, D, D, D, D, D, D, 1, 0, 0, 0, 0, 0, 0);
  // 5) v = mem @ wv_ca + bv_ca            (mem == h)
  gemm(buf_mem, wv_ca, buf_u, bv_ca, nullptr, nullptr, Bn, D, D, D, D, D, 0, 1, 0, 0, 0, 0, 0, 0);
  // 6) h3 = h2 + v @ wo_ca + bo_ca
  gemm(buf_u, wo_ca, buf_h3, bo_ca, nullptr, buf_h2, Bn, D, D, D, D, D, D, 1, 0, 0, 0, 0, 0, 0);
  // 7) t3 = LN3(h3)
  ln(buf_h3, buf_t, ln3_g, ln3_b, D, Bn, Bn);
  // 8) f = gelu(t3 @ w_ff1 + b_ff1)       [Bn, F]
  gemm(buf_t, w_ff1, buf_f, b_ff1, nullptr, nullptr, Bn, F, D, D, F, F, 0, 1, 0, 0, 0, 0, 0, 1);
  // 9) h4 = h3 + f @ w_ff2 + b_ff2
  gemm(buf_f, w_ff2, buf_mem, b_ff2, nullptr, buf_h3, Bn, D, F, F, D, D, D, 1, 0, 0, 0, 0, 0, 0);
  // 10) ho = LNout(h4)
  ln(buf_mem, buf_t, lnout_g, lnout_b, D, Bn, Bn);
  // 11) z[h] = gelu(ho @ wh1[h] + bh1[h])   batched over NH via grid.z
  gemm(buf_t, wh1, buf_z, bh1, nullptr, nullptr, Bn, HD, D, D, HD, HD, 0,
       NH, 0LL, (long long)D * HD, (long long)Bn * HD, (long long)HD, 0LL, 1);
  // 12) zn = LNh(z)  (per-head gamma/beta)
  ln(buf_z, buf_zn, lnh_g, lnh_b, HD, NH * Bn, Bn);
  // 13) logits[:, h, :] = zn[h] @ wh2[h] + bh2[h]  -> d_out [Bn, NH, 896]
  gemm(buf_zn, wh2, out, bh2, nullptr, nullptr, Bn, OUTD, HD, HD, OUTD, NH * OUTD, 0,
       NH, (long long)Bn * HD, (long long)HD * OUTD, (long long)OUTD,
       (long long)OUTD, 0LL, 0);
}
